// OmegaCoreModel_60533269070477
// MI455X (gfx1250) — compile-verified, matrix operand order checked by reading, not run on hardware
//
#include <hip/hip_runtime.h>
#include <math.h>

// ---------------------------------------------------------------------------
// Model dims (compile-time constants from the reference)
// ---------------------------------------------------------------------------
#define T_TOK   1024          // B*L = 2*512
#define SEQ_L   512
#define D_MODEL 1024
#define D_INNER 2048
#define D_STATE 64
#define NHEADS  32
#define HEADDIM 64
#define CONV_DIM 2176         // D_INNER + 2*D_STATE
#define D_IN_PROJ 4256        // 2*D_INNER + 2*D_STATE + NHEADS
#define D_FF    4096
#define N_EXP   8
#define VOCAB   32000
#define EPS_RMS 1e-6f

// ---------------------------------------------------------------------------
// CDNA5 async global->LDS copy (ASYNCcnt-tracked), guarded so the build is
// safe on toolchains that don't declare the builtin.
// Probe-learned signature: (AS1 v4i* gsrc, AS3 v4i* ldsdst, imm offset, imm cpol)
// ---------------------------------------------------------------------------
#if defined(__gfx1250__) && __has_builtin(__builtin_amdgcn_global_load_async_to_lds_b128)
#define HAVE_ASYNC 1
#else
#define HAVE_ASYNC 0
#endif

typedef __attribute__((ext_vector_type(4))) int v4i;

#if HAVE_ASYNC
__device__ __forceinline__ void async_copy_b128(void* lds_dst, const void* g_src) {
  __builtin_amdgcn_global_load_async_to_lds_b128(
      (__attribute__((address_space(1))) v4i*)(g_src),
      (__attribute__((address_space(3))) v4i*)(lds_dst),
      0, 0);
}
#endif

__device__ __forceinline__ void async_wait0() {
#if HAVE_ASYNC
#if __has_builtin(__builtin_amdgcn_s_wait_asynccnt)
  __builtin_amdgcn_s_wait_asynccnt(0);
#else
  asm volatile("s_wait_asynccnt 0x0" ::: "memory");
#endif
#endif
}

// ---------------------------------------------------------------------------
// WMMA types (CDNA5 gfx1250, wave32)
// ---------------------------------------------------------------------------
typedef __attribute__((ext_vector_type(16))) __bf16 v16bf;
typedef __attribute__((ext_vector_type(8)))  float  v8f;

// Pack a 16x32 A/B fragment row (per ISA 7.12.2: lanes 0-15 hold rows,
// half==0 -> K {0..7, 16..23}, half==1 -> K {8..15, 24..31}) from an fp32
// LDS row (stride 32 floats), converting to bf16 in registers.
__device__ __forceinline__ v16bf pack_frag_f32(const float* __restrict__ rowp, int half) {
  v16bf f;
  const float* lo = rowp + half * 8;
  const float* hi = rowp + 16 + half * 8;
#pragma unroll
  for (int i = 0; i < 8; ++i) {
    f[i]     = (__bf16)lo[i];
    f[8 + i] = (__bf16)hi[i];
  }
  return f;
}

// ---------------------------------------------------------------------------
// Tiled WMMA GEMM:  C[M,N] = epi( A[M,K] * W[N,K]^T )
//   epi: +bias[n], optional SiLU, optional * rowscale[m*rs_stride],
//        optional accumulate into existing C (beta)
// Requirements: M % 64 == 0, K % 32 == 0 (true for every call here).
// N is guarded (handles N = 8, 5, ... tails).
// 256 threads = 8 wave32s; tile 64x128, K-step 32.
// Double-buffered fp32 tiles in LDS (48 KB of the 320 KB WGP pool); the
// global->LDS traffic for tile k+1 runs asynchronously (ASYNCcnt) under the
// WMMA work on tile k, retired with s_wait_asynccnt before the swap barrier.
// ---------------------------------------------------------------------------
__global__ void __launch_bounds__(256) wmma_gemm_kernel(
    const float* __restrict__ A, const float* __restrict__ W,
    const float* __restrict__ bias, const float* __restrict__ rowscale, int rs_stride,
    float* __restrict__ C, int M, int N, int K, int act, int beta)
{
  __shared__ float As[2][64 * 32];    // 2 x 8 KB
  __shared__ float Ws[2][128 * 32];   // 2 x 16 KB

  const int tid   = threadIdx.x;
  const int lane  = tid & 31;
  const int wave  = tid >> 5;
  const int half  = lane >> 4;
  const int r16   = lane & 15;
  const int tileM = blockIdx.y * 64;
  const int tileN = blockIdx.x * 128;
  const int waveM = (wave >> 2) * 32;   // 0 / 32
  const int waveN = (wave & 3) * 32;    // 0 / 32 / 64 / 96

  v8f acc[2][2];
#pragma unroll
  for (int mi = 0; mi < 2; ++mi)
#pragma unroll
    for (int ni = 0; ni < 2; ++ni)
#pragma unroll
      for (int e = 0; e < 8; ++e) acc[mi][ni][e] = 0.0f;

  // Stage one 64x32 A tile + 128x32 W tile (fp32) into LDS buffer b.
  auto stage = [&](int b, int k0) {
#pragma unroll
    for (int i = 0; i < 2; ++i) {           // A: 512 x 16B chunks / 256 thr
      int idx = tid + i * 256;
      int r = idx >> 3, c = (idx & 7) * 4;
      const float* gp = A + (size_t)(tileM + r) * K + k0 + c;
      float* lp = &As[b][r * 32 + c];
#if HAVE_ASYNC
      async_copy_b128(lp, gp);
#else
      *(float4*)lp = *(const float4*)gp;
#endif
    }
#pragma unroll
    for (int i = 0; i < 4; ++i) {           // W: 1024 x 16B chunks / 256 thr
      int idx = tid + i * 256;
      int r = idx >> 3, c = (idx & 7) * 4;
      int n = tileN + r;
      float* lp = &Ws[b][r * 32 + c];
      if (n < N) {
        const float* gp = W + (size_t)n * K + k0 + c;
#if HAVE_ASYNC
        async_copy_b128(lp, gp);
#else
        *(float4*)lp = *(const float4*)gp;
#endif
      } else {
        float4 z; z.x = 0.f; z.y = 0.f; z.z = 0.f; z.w = 0.f;
        *(float4*)lp = z;                   // OOB rows: zero-fill (plain store)
      }
    }
  };

  int buf = 0;
  stage(0, 0);
  async_wait0();
  __syncthreads();

  for (int k0 = 0; k0 < K; k0 += 32) {
    // Kick off the next tile's global->LDS copies (async under the WMMAs).
    if (k0 + 32 < K) stage(buf ^ 1, k0 + 32);
    // Cache prefetch two tiles ahead.
    if (k0 + 64 < K) {
      const float* pa = A + (size_t)(tileM + (tid & 63)) * K + (k0 + 64);
      int wr = tileN + (tid >> 1); if (wr >= N) wr = N - 1;
      const float* pw = W + (size_t)wr * K + (k0 + 64);
      __builtin_prefetch(pa, 0, 3);
      __builtin_prefetch(pw, 0, 3);
    }

    v16bf af[2], bw[2];
#pragma unroll
    for (int mi = 0; mi < 2; ++mi)
      af[mi] = pack_frag_f32(&As[buf][(waveM + mi * 16 + r16) * 32], half);
#pragma unroll
    for (int ni = 0; ni < 2; ++ni)
      bw[ni] = pack_frag_f32(&Ws[buf][(waveN + ni * 16 + r16) * 32], half);

#pragma unroll
    for (int mi = 0; mi < 2; ++mi)
#pragma unroll
      for (int ni = 0; ni < 2; ++ni)
        acc[mi][ni] = __builtin_amdgcn_wmma_f32_16x16x32_bf16(
            false, af[mi], false, bw[ni], (short)0, acc[mi][ni], false, false);

    async_wait0();      // next buffer's async copies complete
    __syncthreads();    // everyone done reading buf / writing buf^1
    buf ^= 1;
  }

  // Epilogue per ISA C/D layout: VGPR e, lanes 0-15 -> M=e, lanes 16-31 -> M=e+8.
#pragma unroll
  for (int mi = 0; mi < 2; ++mi)
#pragma unroll
    for (int ni = 0; ni < 2; ++ni)
#pragma unroll
      for (int e = 0; e < 8; ++e) {
        int gr = tileM + waveM + mi * 16 + half * 8 + e;
        int gc = tileN + waveN + ni * 16 + r16;
        if (gc < N) {
          float v = acc[mi][ni][e];
          if (bias) v += bias[gc];
          if (act == 1) v = v / (1.0f + __expf(-v));      // SiLU
          if (rowscale) v *= rowscale[(size_t)gr * rs_stride];
          size_t off = (size_t)gr * N + gc;
          C[off] = beta ? (C[off] + v) : v;
        }
      }
}

// ---------------------------------------------------------------------------
// Small kernels
// ---------------------------------------------------------------------------
__global__ void zero_kernel(float* p, int n) {
  int i = blockIdx.x * blockDim.x + threadIdx.x;
  if (i < n) p[i] = 0.0f;
}

__global__ void embed_gather_kernel(const int* __restrict__ ids,
                                    const float* __restrict__ embed,
                                    float* __restrict__ x) {
  int t = blockIdx.x;
  int id = ids[t];
  const float* src = embed + (size_t)id * D_MODEL;
  float* dst = x + (size_t)t * D_MODEL;
  for (int d = threadIdx.x; d < D_MODEL; d += blockDim.x) dst[d] = src[d];
}

__global__ void rmsnorm_kernel(const float* __restrict__ x,
                               const float* __restrict__ w,
                               float* __restrict__ out, int D) {
  __shared__ float red[256];
  int t = blockIdx.x;
  const float* xr = x + (size_t)t * D;
  float ss = 0.f;
  for (int d = threadIdx.x; d < D; d += 256) { float v = xr[d]; ss += v * v; }
  red[threadIdx.x] = ss;
  __syncthreads();
  for (int s = 128; s > 0; s >>= 1) {
    if (threadIdx.x < s) red[threadIdx.x] += red[threadIdx.x + s];
    __syncthreads();
  }
  float scale = rsqrtf(red[0] / (float)D + EPS_RMS);
  for (int d = threadIdx.x; d < D; d += 256) out[(size_t)t * D + d] = xr[d] * scale * w[d];
}

// Depthwise causal conv (D_CONV=4) over the xBC slice of zxbcdt + SiLU.
__global__ void conv_silu_kernel(const float* __restrict__ zx,
                                 const float* __restrict__ cw,
                                 const float* __restrict__ cb,
                                 float* __restrict__ out) {
  int idx = blockIdx.x * blockDim.x + threadIdx.x;
  if (idx >= T_TOK * CONV_DIM) return;
  int t = idx / CONV_DIM, c = idx % CONV_DIM;
  int b = t >> 9, l = t & 511;
  float acc = cb[c];
#pragma unroll
  for (int j = 0; j < 4; ++j) {
    int ls = l - 3 + j;
    if (ls >= 0)
      acc += cw[j * CONV_DIM + c] * zx[(size_t)(b * SEQ_L + ls) * D_IN_PROJ + D_INNER + c];
  }
  out[(size_t)t * CONV_DIM + c] = acc / (1.f + __expf(-acc));
}

// Sequential SSM scan: one block per (batch, head); 64 threads own HEADDIM
// rows of the state; 64-wide state kept in registers.
__global__ void scan_kernel(const float* __restrict__ zx,
                            const float* __restrict__ xBC,
                            const float* __restrict__ A_log,
                            const float* __restrict__ dt_bias,
                            const float* __restrict__ Dp,
                            float* __restrict__ ybuf) {
  int b = blockIdx.x >> 5;   // NHEADS = 32
  int h = blockIdx.x & 31;
  int p = threadIdx.x;       // 0..63
  __shared__ float Bs[D_STATE], Cs[D_STATE];
  float hst[D_STATE];
#pragma unroll
  for (int n = 0; n < D_STATE; ++n) hst[n] = 0.f;
  float A   = -__expf(A_log[h]);
  float dtb = dt_bias[h];
  float Dh  = Dp[h];
  for (int l = 0; l < SEQ_L; ++l) {
    int t = b * SEQ_L + l;
    const float* xrow = xBC + (size_t)t * CONV_DIM;
    Bs[p] = xrow[D_INNER + p];
    Cs[p] = xrow[D_INNER + D_STATE + p];
    float dtraw = zx[(size_t)t * D_IN_PROJ + (D_IN_PROJ - NHEADS) + h] + dtb;
    float dt = (dtraw > 20.f) ? dtraw : __logf(1.f + __expf(dtraw));  // softplus
    __syncthreads();
    float dA = __expf(dt * A);
    float xs = xrow[h * HEADDIM + p];
    float dx = dt * xs;
    float y = 0.f;
#pragma unroll
    for (int n = 0; n < D_STATE; ++n) {
      hst[n] = hst[n] * dA + dx * Bs[n];
      y += hst[n] * Cs[n];
    }
    ybuf[(size_t)t * D_INNER + h * HEADDIM + p] = y + Dh * xs;
    __syncthreads();
  }
}

// y' = rmsnorm(y * silu(z)) * w  over D_INNER; z lives in zxbcdt[:, :D_INNER].
__global__ void gated_rmsnorm_kernel(const float* __restrict__ y,
                                     const float* __restrict__ zx,
                                     const float* __restrict__ w,
                                     float* __restrict__ out) {
  __shared__ float red[256];
  int t = blockIdx.x;
  float v[8];
  float ss = 0.f;
#pragma unroll
  for (int i = 0; i < 8; ++i) {
    int d = threadIdx.x + i * 256;
    float zf = zx[(size_t)t * D_IN_PROJ + d];
    float yv = y[(size_t)t * D_INNER + d];
    float sv = yv * (zf / (1.f + __expf(-zf)));
    v[i] = sv;
    ss += sv * sv;
  }
  red[threadIdx.x] = ss;
  __syncthreads();
  for (int s = 128; s > 0; s >>= 1) {
    if (threadIdx.x < s) red[threadIdx.x] += red[threadIdx.x + s];
    __syncthreads();
  }
  float scale = rsqrtf(red[0] / (float)D_INNER + EPS_RMS);
#pragma unroll
  for (int i = 0; i < 8; ++i) {
    int d = threadIdx.x + i * 256;
    out[(size_t)t * D_INNER + d] = v[i] * scale * w[d];
  }
}

// Top-2 softmax routing + full-softmax column sums (for the LB loss).
__global__ void moe_gate_kernel(const float* __restrict__ gl,
                                float* __restrict__ we,
                                float* __restrict__ gp_sum, int T) {
  int t = blockIdx.x * blockDim.x + threadIdx.x;
  if (t >= T) return;
  float g[N_EXP];
  float gmax = -1e30f;
#pragma unroll
  for (int e = 0; e < N_EXP; ++e) { g[e] = gl[t * N_EXP + e]; gmax = fmaxf(gmax, g[e]); }
  float Z = 0.f, p[N_EXP];
#pragma unroll
  for (int e = 0; e < N_EXP; ++e) { p[e] = __expf(g[e] - gmax); Z += p[e]; }
#pragma unroll
  for (int e = 0; e < N_EXP; ++e) atomicAdd(&gp_sum[e], p[e] / Z);
  int i1 = 0;
#pragma unroll
  for (int e = 1; e < N_EXP; ++e) if (g[e] > g[i1]) i1 = e;
  int i2 = -1;
#pragma unroll
  for (int e = 0; e < N_EXP; ++e) if (e != i1 && (i2 < 0 || g[e] > g[i2])) i2 = e;
  float e1 = 1.f, e2 = __expf(g[i2] - g[i1]);
  float s = e1 + e2;
#pragma unroll
  for (int e = 0; e < N_EXP; ++e) we[t * N_EXP + e] = 0.f;
  we[t * N_EXP + i1] = e1 / s;
  we[t * N_EXP + i2] = e2 / s;
}

__global__ void moe_lb_kernel(const float* __restrict__ gp_sum, float* lb, float Tf) {
  if (blockIdx.x == 0 && threadIdx.x == 0) {
    float s = 0.f;
#pragma unroll
    for (int e = 0; e < N_EXP; ++e) { float m = gp_sum[e] / Tf; s += m * m; }
    *lb += (float)N_EXP * s;
  }
}

// Safety heads on the 128-wide feature f; layer-weighted accumulation.
__global__ void safety_head_kernel(const float* __restrict__ f2,
                                   const float* __restrict__ harm_w, const float* __restrict__ harm_b,
                                   const float* __restrict__ unc_w,  const float* __restrict__ unc_b,
                                   const float* __restrict__ tox_w,  const float* __restrict__ tox_b,
                                   float* __restrict__ risk_out, float* __restrict__ unc_out,
                                   float* __restrict__ tox_out, float wt, int T) {
  int t = blockIdx.x * blockDim.x + threadIdx.x;
  if (t >= T) return;
  const float* f = f2 + (size_t)t * 128;
  float risk = 0.f;
#pragma unroll
  for (int j = 0; j < 5; ++j) {
    float a = harm_b[j];
    for (int d = 0; d < 128; ++d) a += f[d] * harm_w[j * 128 + d];
    risk = fmaxf(risk, 1.f / (1.f + __expf(-a)));
  }
  float u = unc_b[0];
  for (int d = 0; d < 128; ++d) u += f[d] * unc_w[d];
  u = 1.f / (1.f + __expf(-u));
  float x = tox_b[0];
  for (int d = 0; d < 128; ++d) x += f[d] * tox_w[d];
  x = 1.f / (1.f + __expf(-x));
  risk_out[t] += wt * risk;
  unc_out[t]  += wt * u;
  tox_out[t]  += wt * x;
}

// ---------------------------------------------------------------------------
// Host orchestration
// ---------------------------------------------------------------------------
struct SafetyP {
  const float *b1, *b2, *harm_b, *harm_w, *tox_b, *tox_w, *unc_b, *unc_w, *w1, *w2;
};
struct LayerP {
  const float *A_log, *D, *conv_b, *conv_w, *dt_bias, *gate, *in_proj;
  const float *norm_gated, *norm_mamba, *norm_moe, *out_proj, *w1, *w2;
  SafetyP s;
  bool moe;
};

static void launch_gemm(const float* A, const float* W, const float* bias,
                        const float* rowscale, int rs_stride, float* C,
                        int M, int N, int K, int act, int beta, hipStream_t s) {
  dim3 grid((N + 127) / 128, (M + 63) / 64);
  wmma_gemm_kernel<<<grid, dim3(256), 0, s>>>(A, W, bias, rowscale, rs_stride, C,
                                              M, N, K, act, beta);
}

extern "C" void kernel_launch(void* const* d_in, const int* in_sizes, int n_in,
                              void* d_out, int out_size, void* d_ws, size_t ws_size,
                              hipStream_t stream) {
  (void)in_sizes; (void)n_in; (void)out_size; (void)ws_size;

  // ---- inputs (jax pytree leaves in sorted-key order) ----
  const int*   ids   = (const int*)d_in[0];
  const float* embed = (const float*)d_in[1];
  int idx = 2;
  auto F = [&]() -> const float* { return (const float*)d_in[idx++]; };

  LayerP L[4];
  for (int i = 0; i < 4; ++i) {
    bool moe = (i % 2 == 0);
    LayerP& lp = L[i];
    lp.moe = moe;
    lp.A_log = F(); lp.D = F(); lp.conv_b = F(); lp.conv_w = F(); lp.dt_bias = F();
    lp.gate = nullptr;
    if (moe) lp.gate = F();
    lp.in_proj = F(); lp.norm_gated = F(); lp.norm_mamba = F();
    lp.norm_moe = nullptr;
    if (moe) lp.norm_moe = F();
    lp.out_proj = F();
    lp.s.b1 = F(); lp.s.b2 = F(); lp.s.harm_b = F(); lp.s.harm_w = F();
    lp.s.tox_b = F(); lp.s.tox_w = F(); lp.s.unc_b = F(); lp.s.unc_w = F();
    lp.s.w1 = F(); lp.s.w2 = F();
    lp.w1 = nullptr; lp.w2 = nullptr;
    if (moe) { lp.w1 = F(); lp.w2 = F(); }
  }
  const float* norm_f = F();

  // ---- workspace layout (fp32) ----
  float* cur = (float*)d_ws;
  auto alloc = [&](size_t n) -> float* { float* p = cur; cur += n; return p; };
  float* x      = alloc((size_t)T_TOK * D_MODEL);
  float* xn     = alloc((size_t)T_TOK * D_MODEL);
  float* zx     = alloc((size_t)T_TOK * D_IN_PROJ);
  float* xbc    = alloc((size_t)T_TOK * CONV_DIM);
  float* ybuf   = alloc((size_t)T_TOK * D_INNER);
  float* gated  = alloc((size_t)T_TOK * D_INNER);
  float* gl     = alloc((size_t)T_TOK * N_EXP);
  float* we     = alloc((size_t)T_TOK * N_EXP);
  float* he     = alloc((size_t)T_TOK * D_FF);
  float* f1     = alloc((size_t)T_TOK * 512);
  float* f2     = alloc((size_t)T_TOK * 128);
  float* gp_sum = alloc(8);

  // ---- output layout: logits | risk | unc | tox | lb ----
  float* logits = (float*)d_out;
  float* risk = logits + (size_t)T_TOK * VOCAB;
  float* unc  = risk + T_TOK;
  float* tox  = unc + T_TOK;
  float* lb   = tox + T_TOK;

  zero_kernel<<<(3 * T_TOK + 1 + 255) / 256, 256, 0, stream>>>(risk, 3 * T_TOK + 1);

  // wts = softmax(linspace(0,1,4))
  float ew[4], esum = 0.f, wts[4];
  for (int i = 0; i < 4; ++i) { ew[i] = expf((float)i / 3.0f); esum += ew[i]; }
  for (int i = 0; i < 4; ++i) wts[i] = ew[i] / esum;

  embed_gather_kernel<<<T_TOK, 256, 0, stream>>>(ids, embed, x);

  for (int i = 0; i < 4; ++i) {
    const LayerP& lp = L[i];
    // ---- Mamba2 block ----
    rmsnorm_kernel<<<T_TOK, 256, 0, stream>>>(x, lp.norm_mamba, xn, D_MODEL);
    launch_gemm(xn, lp.in_proj, nullptr, nullptr, 0, zx,
                T_TOK, D_IN_PROJ, D_MODEL, 0, 0, stream);
    conv_silu_kernel<<<(T_TOK * CONV_DIM + 255) / 256, 256, 0, stream>>>(
        zx, lp.conv_w, lp.conv_b, xbc);
    scan_kernel<<<2 * NHEADS, HEADDIM, 0, stream>>>(zx, xbc, lp.A_log, lp.dt_bias, lp.D, ybuf);
    gated_rmsnorm_kernel<<<T_TOK, 256, 0, stream>>>(ybuf, zx, lp.norm_gated, gated);
    // residual: x += gated @ out_proj^T   (beta = 1)
    launch_gemm(gated, lp.out_proj, nullptr, nullptr, 0, x,
                T_TOK, D_MODEL, D_INNER, 0, 1, stream);

    // ---- MoE block (layers 0, 2) ----
    if (lp.moe) {
      rmsnorm_kernel<<<T_TOK, 256, 0, stream>>>(x, lp.norm_moe, xn, D_MODEL);
      launch_gemm(xn, lp.gate, nullptr, nullptr, 0, gl,
                  T_TOK, N_EXP, D_MODEL, 0, 0, stream);
      zero_kernel<<<1, 256, 0, stream>>>(gp_sum, 8);
      moe_gate_kernel<<<(T_TOK + 255) / 256, 256, 0, stream>>>(gl, we, gp_sum, T_TOK);
      moe_lb_kernel<<<1, 1, 0, stream>>>(gp_sum, lb, (float)T_TOK);
      for (int e = 0; e < N_EXP; ++e) {
        // h_e = silu(xn @ w1_e^T)
        launch_gemm(xn, lp.w1 + (size_t)e * D_FF * D_MODEL, nullptr, nullptr, 0, he,
                    T_TOK, D_FF, D_MODEL, 1, 0, stream);
        // x += we[:,e] * (h_e @ w2_e^T)
        launch_gemm(he, lp.w2 + (size_t)e * D_MODEL * D_FF, nullptr, we + e, N_EXP, x,
                    T_TOK, D_MODEL, D_FF, 0, 1, stream);
      }
    }

    // ---- safety guard (on raw residual x) ----
    launch_gemm(x, lp.s.w1, lp.s.b1, nullptr, 0, f1, T_TOK, 512, D_MODEL, 1, 0, stream);
    launch_gemm(f1, lp.s.w2, lp.s.b2, nullptr, 0, f2, T_TOK, 128, 512, 0, 0, stream);
    safety_head_kernel<<<(T_TOK + 255) / 256, 256, 0, stream>>>(
        f2, lp.s.harm_w, lp.s.harm_b, lp.s.unc_w, lp.s.unc_b, lp.s.tox_w, lp.s.tox_b,
        risk, unc, tox, wts[i], T_TOK);
  }

  // ---- final norm + logits ----
  rmsnorm_kernel<<<T_TOK, 256, 0, stream>>>(x, norm_f, xn, D_MODEL);
  launch_gemm(xn, embed, nullptr, nullptr, 0, logits, T_TOK, VOCAB, D_MODEL, 0, 0, stream);
}
